// QueST_48885317763708
// MI455X (gfx1250) — matrix operand order repaired and checked by reading, not compiled, not run on hardware
//
#include <hip/hip_runtime.h>

// ---------------------------------------------------------------------------
// Types for CDNA5 WMMA (wave32)
// ---------------------------------------------------------------------------
typedef __attribute__((ext_vector_type(16))) __bf16 v16bf;
typedef __attribute__((ext_vector_type(8)))  __bf16 v8bf;
typedef __attribute__((ext_vector_type(8)))  float  v8f;

static __device__ inline v16bf cat16(v8bf lo, v8bf hi) {
    union { v16bf v; v8bf h[2]; } u;
    u.h[0] = lo; u.h[1] = hi;
    return u.v;
}

// Async global->LDS copy of 32 contiguous bytes (two B128 transfers).
// lds    = workgroup-relative LDS byte address (low 32 bits of flat shared ptr)
// gaddr  = 64-bit global address
// Tracked by ASYNCcnt; INST_OFFSET applies to both LDS and memory address.
static __device__ inline void async_copy32(unsigned lds, const void* gptr) {
    unsigned long long ga = (unsigned long long)(size_t)gptr;
    asm volatile("global_load_async_to_lds_b128 %0, %1, off\n\t"
                 "global_load_async_to_lds_b128 %0, %1, off offset:16"
                 :: "v"(lds), "v"(ga) : "memory");
}

static __device__ inline void wait_async0() {
    asm volatile("s_wait_asynccnt 0x0" ::: "memory");
}

// ---------------------------------------------------------------------------
// WMMA GEMM:  C[M,N] = act( A[M,K] @ W[N,K]^T + bias[N] )
//   A, W : bf16, row-major, K contiguous (weights pre-packed to bf16 in ws)
//   Block tile 128x128x32, 256 threads = 8 wave32 waves, wave tile 64x32.
//   Tiles staged with GLOBAL_LOAD_ASYNC_TO_LDS_B128, double buffered.
//   ACT: 0 = identity, 1 = ReLU
// ---------------------------------------------------------------------------
#define BM 128
#define BN 128
#define BK 32
#define LDSK 40   // BK + 8 pad -> 80B row stride (16B aligned for b128 DS ops)

template <int ACT>
__global__ __launch_bounds__(256) void wmma_gemm_bf16(
    const __bf16* __restrict__ A, int K,
    const __bf16* __restrict__ W,
    const float* __restrict__ bias,
    float* __restrict__ C, int N)
{
    __shared__ alignas(16) __bf16 As[2][BM][LDSK];
    __shared__ alignas(16) __bf16 Bs[2][BN][LDSK];

    const int tid = threadIdx.x;
    const int m0  = blockIdx.y * BM;
    const int n0  = blockIdx.x * BN;

    // ---- async staging: 2 threads per row, 16 bf16 (32B) each -------------
    const int ldRow  = tid >> 1;                  // 0..127
    const int ldHalf = tid & 1;                   // 16-element K half
    const __bf16* aG = A + (size_t)(m0 + ldRow) * K + ldHalf * 16;
    const __bf16* wG = W + (size_t)(n0 + ldRow) * K + ldHalf * 16;

    const unsigned bufStride = BM * LDSK * sizeof(__bf16);   // 10240 B
    const unsigned ldsA = (unsigned)(size_t)&As[0][ldRow][ldHalf * 16];
    const unsigned ldsB = (unsigned)(size_t)&Bs[0][ldRow][ldHalf * 16];

    auto stage = [&](int buf, int k0) {
        const unsigned o = (unsigned)buf * bufStride;
        async_copy32(ldsA + o, aG + k0);
        async_copy32(ldsB + o, wG + k0);
    };

    // ---- wave geometry ----------------------------------------------------
    const int wave = tid >> 5;
    const int wm   = wave >> 2;      // 0..1  -> 64-row half
    const int wn   = wave & 3;       // 0..3  -> 32-col quarter
    const int lane = tid & 31;
    const int lr   = lane & 15;
    const int lh   = lane >> 4;      // lane half per ISA fragment layout

    v8f acc[4][2];
    #pragma unroll
    for (int i = 0; i < 4; ++i)
        #pragma unroll
        for (int j = 0; j < 2; ++j)
            acc[i][j] = (v8f)(0.0f);

    const int aRowBase = wm * 64 + lr;   // + fm*16
    const int aKlo     = lh * 8;         // A frag: K {0-7,16-23} / {8-15,24-31}
    const int bColBase = wn * 32 + lr;   // + fn*16
    const int bK       = lh * 8;         // B frag: K {0-7,16-23} / {8-15,24-31}

    stage(0, 0);
    wait_async0();
    __syncthreads();

    const int steps = K / BK;
    for (int it = 0; it < steps; ++it) {
        const int buf = it & 1;
        if (it + 1 < steps) stage(buf ^ 1, (it + 1) * BK);

        v16bf afr[4];
        #pragma unroll
        for (int fm = 0; fm < 4; ++fm) {
            const __bf16* p = &As[buf][aRowBase + fm * 16][0];
            afr[fm] = cat16(*(const v8bf*)(p + aKlo),
                            *(const v8bf*)(p + aKlo + 16));
        }
        v16bf bfr[2];
        #pragma unroll
        for (int fn = 0; fn < 2; ++fn) {
            const __bf16* p = &Bs[buf][bColBase + fn * 16][bK];
            bfr[fn] = cat16(*(const v8bf*)(p),
                            *(const v8bf*)(p + 16));
        }
        #pragma unroll
        for (int fm = 0; fm < 4; ++fm)
            #pragma unroll
            for (int fn = 0; fn < 2; ++fn)
                acc[fm][fn] = __builtin_amdgcn_wmma_f32_16x16x32_bf16(
                    false, afr[fm], false, bfr[fn], (short)0, acc[fm][fn],
                    false, false);

        wait_async0();        // this wave's copies into buf^1 are done
        __syncthreads();      // everyone's copies visible before next read
    }

    // ---- epilogue: bias + activation, f32 store ---------------------------
    // C/D layout: VGPR r -> rows M=r (lanes 0-15) / M=r+8 (lanes 16-31), N=lr
    #pragma unroll
    for (int fm = 0; fm < 4; ++fm) {
        #pragma unroll
        for (int fn = 0; fn < 2; ++fn) {
            const int col  = n0 + wn * 32 + fn * 16 + lr;
            const float bv = bias[col];
            #pragma unroll
            for (int r = 0; r < 8; ++r) {
                const int row = m0 + wm * 64 + fm * 16 + lh * 8 + r;
                float v = acc[fm][fn][r] + bv;
                if (ACT == 1) v = fmaxf(v, 0.0f);
                C[(size_t)row * N + col] = v;
            }
        }
    }
}

// ---------------------------------------------------------------------------
// Pack kernels: f32 -> bf16, 8 elements per thread (all sizes are /8)
// ---------------------------------------------------------------------------
__global__ void pack_bf16x8(const float* __restrict__ A,
                            __bf16* __restrict__ Z, int total8)
{
    int i = blockIdx.x * blockDim.x + threadIdx.x;
    if (i >= total8) return;
    const float4* p = (const float4*)(A + (size_t)i * 8);
    float4 a = p[0], b = p[1];
    v8bf o;
    o[0] = (__bf16)a.x; o[1] = (__bf16)a.y; o[2] = (__bf16)a.z; o[3] = (__bf16)a.w;
    o[4] = (__bf16)b.x; o[5] = (__bf16)b.y; o[6] = (__bf16)b.z; o[7] = (__bf16)b.w;
    *(v8bf*)(Z + (size_t)i * 8) = o;
}

// Z[row, 0:k1) = A[row], Z[row, k1:k1+k2) = Bm[row]; k1,k2 multiples of 8
__global__ void pack_concat_bf16x8(const float* __restrict__ A,
                                   const float* __restrict__ Bm,
                                   __bf16* __restrict__ Z,
                                   int k1, int k2, int total8)
{
    int i = blockIdx.x * blockDim.x + threadIdx.x;
    if (i >= total8) return;
    const int Kt  = k1 + k2;
    const int Kt8 = Kt / 8;
    const int row = i / Kt8;
    const int col = (i - row * Kt8) * 8;
    const float* src = (col < k1) ? (A  + (size_t)row * k1 + col)
                                  : (Bm + (size_t)row * k2 + (col - k1));
    float4 a = ((const float4*)src)[0];
    float4 b = ((const float4*)src)[1];
    v8bf o;
    o[0] = (__bf16)a.x; o[1] = (__bf16)a.y; o[2] = (__bf16)a.z; o[3] = (__bf16)a.w;
    o[4] = (__bf16)b.x; o[5] = (__bf16)b.y; o[6] = (__bf16)b.z; o[7] = (__bf16)b.w;
    *(v8bf*)(Z + (size_t)row * Kt + col) = o;
}

// dst[g*n + j] = pg[j]   (concatenated gate biases, f32)
__global__ void concat4_f32(float* __restrict__ dst,
                            const float* __restrict__ p0, const float* __restrict__ p1,
                            const float* __restrict__ p2, const float* __restrict__ p3,
                            int n)
{
    int i = blockIdx.x * blockDim.x + threadIdx.x;
    if (i >= 4 * n) return;
    const int g = i / n, j = i - g * n;
    const float* s = (g == 0) ? p0 : (g == 1) ? p1 : (g == 2) ? p2 : p3;
    dst[i] = s[j];
}

// ---------------------------------------------------------------------------
// LSTM pointwise (faithful to reference, incl. the double tanh):
//   f=sig(Gf) i=sig(Gi) cand=tanh(Gc) o=sig(Go)
//   h = o * tanh(cand) ; c = f*c_prev + cand*i
// G is [B, 4*nh] with gate order f,i,c,o along columns.
// ---------------------------------------------------------------------------
static __device__ inline float sigmoid_(float x) {
    return 1.0f / (1.0f + __expf(-x));
}

__global__ void lstm_pointwise(const float* __restrict__ G,
                               const float* __restrict__ cprev,
                               float* __restrict__ h,
                               float* __restrict__ c,
                               int nh, int total)
{
    int idx = blockIdx.x * blockDim.x + threadIdx.x;
    if (idx >= total) return;
    const int row = idx / nh;
    const int col = idx - row * nh;
    const float* gr = G + (size_t)row * 4 * nh;
    const float f    = sigmoid_(gr[col]);
    const float i    = sigmoid_(gr[nh + col]);
    const float cand = tanhf(gr[2 * nh + col]);
    const float o    = sigmoid_(gr[3 * nh + col]);
    h[idx] = o * tanhf(cand);
    c[idx] = f * cprev[idx] + cand * i;
}

// ---------------------------------------------------------------------------
// Launch sequence
// ---------------------------------------------------------------------------
extern "C" void kernel_launch(void* const* d_in, const int* in_sizes, int n_in,
                              void* d_out, int out_size, void* d_ws, size_t ws_size,
                              hipStream_t stream) {
    constexpr int Bsz = 4096, NI = 1024, NH1 = 1024, NH2 = 1024, NH3 = 2048, NH4 = 1024;

    const float* X    = (const float*)d_in[0];
    const float* h1p  = (const float*)d_in[1];
    const float* h2p  = (const float*)d_in[2];
    const float* c1p  = (const float*)d_in[3];
    const float* c2p  = (const float*)d_in[4];
    const float* Wf1 = (const float*)d_in[5];  const float* bf1v = (const float*)d_in[6];
    const float* Wi1 = (const float*)d_in[7];  const float* bi1  = (const float*)d_in[8];
    const float* Wc1 = (const float*)d_in[9];  const float* bc1  = (const float*)d_in[10];
    const float* Wo1 = (const float*)d_in[11]; const float* bo1  = (const float*)d_in[12];
    const float* Wf2 = (const float*)d_in[13]; const float* bf2  = (const float*)d_in[14];
    const float* Wi2 = (const float*)d_in[15]; const float* bi2  = (const float*)d_in[16];
    const float* Wc2 = (const float*)d_in[17]; const float* bc2  = (const float*)d_in[18];
    const float* Wo2 = (const float*)d_in[19]; const float* bo2  = (const float*)d_in[20];
    const float* W3  = (const float*)d_in[21]; const float* b3   = (const float*)d_in[22];
    const float* W4  = (const float*)d_in[23]; const float* b4   = (const float*)d_in[24];

    float* out = (float*)d_out;
    const size_t S = (size_t)Bsz * 1024;
    float* h1 = out + S;
    float* h2 = out + 2 * S;
    float* c1 = out + 3 * S;
    float* c2 = out + 4 * S;

    // ws layout (bytes):
    //   [  0MB, 16MB)  Z      bf16 activations (4096 x 2048)
    //   [ 16MB, 80MB)  G      f32 gate pre-activations (4096 x 4096) / out3
    //   [ 80MB, 96MB)  Wcat1  bf16 [4096 x 2048] = [Wf1;Wi1;Wc1;Wo1]
    //   [ 96MB,112MB)  Wcat2  bf16 [4096 x 2048]
    //   [112MB,116MB)  W3b    bf16 [2048 x 1024]
    //   [116MB,120MB)  W4b    bf16 [1024 x 2048]
    //   [120MB, ...)   bcat1, bcat2 f32 [4096] each
    char* ws = (char*)d_ws;
    const size_t MB = 1024 * 1024;
    __bf16* Z     = (__bf16*)ws;
    float*  G     = (float*)(ws + 16 * MB);
    __bf16* Wcat1 = (__bf16*)(ws + 80 * MB);
    __bf16* Wcat2 = (__bf16*)(ws + 96 * MB);
    __bf16* W3b   = (__bf16*)(ws + 112 * MB);
    __bf16* W4b   = (__bf16*)(ws + 116 * MB);
    float*  bcat1 = (float*)(ws + 120 * MB);
    float*  bcat2 = bcat1 + 4096;
    float*  out3  = G;   // reuses G region after layer-2 gates are consumed

    const int tW   = NH1 * (NI + NH1);      // 2M elems per gate weight
    const int tZ2  = Bsz * 2048;
    const int tH   = Bsz * 1024;
    const int tO3  = Bsz * NH3;
    dim3 blk(256);
    auto g8 = [](int n) { return dim3((n / 8 + 255) / 256); };

    // ---- one-time weight repack to bf16 (stream-ordered, deterministic) ---
    pack_bf16x8<<<g8(tW), blk, 0, stream>>>(Wf1, Wcat1 + (size_t)0 * tW, tW / 8);
    pack_bf16x8<<<g8(tW), blk, 0, stream>>>(Wi1, Wcat1 + (size_t)1 * tW, tW / 8);
    pack_bf16x8<<<g8(tW), blk, 0, stream>>>(Wc1, Wcat1 + (size_t)2 * tW, tW / 8);
    pack_bf16x8<<<g8(tW), blk, 0, stream>>>(Wo1, Wcat1 + (size_t)3 * tW, tW / 8);
    pack_bf16x8<<<g8(tW), blk, 0, stream>>>(Wf2, Wcat2 + (size_t)0 * tW, tW / 8);
    pack_bf16x8<<<g8(tW), blk, 0, stream>>>(Wi2, Wcat2 + (size_t)1 * tW, tW / 8);
    pack_bf16x8<<<g8(tW), blk, 0, stream>>>(Wc2, Wcat2 + (size_t)2 * tW, tW / 8);
    pack_bf16x8<<<g8(tW), blk, 0, stream>>>(Wo2, Wcat2 + (size_t)3 * tW, tW / 8);
    pack_bf16x8<<<g8(NH3 * NH2), blk, 0, stream>>>(W3, W3b, NH3 * NH2 / 8);
    pack_bf16x8<<<g8(NH4 * NH3), blk, 0, stream>>>(W4, W4b, NH4 * NH3 / 8);
    concat4_f32<<<dim3(16), blk, 0, stream>>>(bcat1, bf1v, bi1, bc1, bo1, NH1);
    concat4_f32<<<dim3(16), blk, 0, stream>>>(bcat2, bf2, bi2, bc2, bo2, NH2);

    // ---- LSTM layer 1 -----------------------------------------------------
    pack_concat_bf16x8<<<g8(tZ2), blk, 0, stream>>>(X, h1p, Z, NI, NH1, tZ2 / 8);
    wmma_gemm_bf16<0><<<dim3(4 * NH1 / BN, Bsz / BM), blk, 0, stream>>>(
        Z, NI + NH1, Wcat1, bcat1, G, 4 * NH1);
    lstm_pointwise<<<dim3((tH + 255) / 256), blk, 0, stream>>>(G, c1p, h1, c1, NH1, tH);

    // ---- LSTM layer 2 -----------------------------------------------------
    pack_concat_bf16x8<<<g8(tZ2), blk, 0, stream>>>(h1, h2p, Z, NH1, NH2, tZ2 / 8);
    wmma_gemm_bf16<0><<<dim3(4 * NH2 / BN, Bsz / BM), blk, 0, stream>>>(
        Z, NH1 + NH2, Wcat2, bcat2, G, 4 * NH2);
    lstm_pointwise<<<dim3((tH + 255) / 256), blk, 0, stream>>>(G, c2p, h2, c2, NH2, tH);

    // ---- MLP head ---------------------------------------------------------
    pack_bf16x8<<<g8(tH), blk, 0, stream>>>(h2, Z, tH / 8);
    wmma_gemm_bf16<1><<<dim3(NH3 / BN, Bsz / BM), blk, 0, stream>>>(
        Z, NH2, W3b, b3, out3, NH3);
    pack_bf16x8<<<g8(tO3), blk, 0, stream>>>(out3, Z, tO3 / 8);
    wmma_gemm_bf16<0><<<dim3(NH4 / BN, Bsz / BM), blk, 0, stream>>>(
        Z, NH3, W4b, b4, out, NH4);

    (void)in_sizes; (void)n_in; (void)out_size; (void)ws_size;
}
